// LacssModel_33105607918282
// MI455X (gfx1250) — matrix operand order repaired and checked by reading, not compiled, not run on hardware
//
#include <hip/hip_runtime.h>
#include <hip/hip_bf16.h>

typedef __attribute__((ext_vector_type(16))) _Float16 v16h;
typedef __attribute__((ext_vector_type(8)))  _Float16 v8h;
typedef __attribute__((ext_vector_type(8)))  float    v8f;

union AV { v16h v; v8h h[2]; };

#define BATCH 4
#define HF 128
#define WF 128
#define K_PRE 2000
#define K_OUT 500
#define PS 32

// ---------------- weight repack: HWIO f32 -> [N][Kpad] f16 (zero-padded K) ----------------
__global__ void repack_wT(const float* __restrict__ w, _Float16* __restrict__ bt,
                          int K, int N, int Kpad) {
    int idx = blockIdx.x * 256 + threadIdx.x;
    if (idx >= N * Kpad) return;
    int n = idx / Kpad, k = idx % Kpad;
    float v = (k < K) ? w[(size_t)k * N + n] : 0.f;
    bt[(size_t)n * Kpad + k] = (_Float16)v;
}

// ---------------- im2col for conv1 (512 -> 256, stride 2, SAME pad lo=0 hi=1) -------------
__global__ void im2col_conv1(const float* __restrict__ img, _Float16* __restrict__ a1) {
    int m = blockIdx.x * 256 + threadIdx.x;
    if (m >= BATCH * 256 * 256) return;
    int x = m & 255, y = (m >> 8) & 255, b = m >> 16;
    _Float16* row = a1 + (size_t)m * 32;
    int kk = 0;
    for (int ky = 0; ky < 3; ++ky)
        for (int kx = 0; kx < 3; ++kx) {
            int iy = 2 * y + ky, ix = 2 * x + kx;
            bool ok = (iy < 512) && (ix < 512);
            const float* p = img + (((size_t)(b * 512 + iy)) * 512 + ix) * 3;
            for (int c = 0; c < 3; ++c) {
                row[kk] = ok ? (_Float16)p[c] : (_Float16)0.f;
                ++kk;
            }
        }
    for (; kk < 32; ++kk) row[kk] = (_Float16)0.f;
}

// ---------------- im2col for conv2 (256 -> 128, stride 2, SAME pad lo=0 hi=1) -------------
__global__ void im2col_conv2(const _Float16* __restrict__ X, _Float16* __restrict__ A2) {
    int idx = blockIdx.x * 256 + threadIdx.x;
    if (idx >= BATCH * HF * WF * 9) return;
    int tap = idx % 9;
    int m = idx / 9;
    int x = m & 127, y = (m >> 7) & 127, b = m >> 14;
    int ky = tap / 3, kx = tap % 3;
    int iy = 2 * y + ky, ix = 2 * x + kx;
    _Float16* dst = A2 + (size_t)m * 576 + tap * 64;
    if (iy < 256 && ix < 256) {
        const _Float16* src = X + (((size_t)(b * 256 + iy)) * 256 + ix) * 64;
        for (int i = 0; i < 8; ++i) ((v8h*)dst)[i] = ((const v8h*)src)[i];
    } else {
        for (int i = 0; i < 64; ++i) dst[i] = (_Float16)0.f;
    }
}

// ---------------- generic WMMA GEMM: C = relu(A[M,K]*Bt[N,K]^T + bias), f16 out -----------
// block = 128 threads = 4 waves; each wave -> FOUR 16-row M tiles x 32 cols (B reuse 4x).
// Requires M % 256 == 0, N % 32 == 0, K % 32 == 0.
__global__ __launch_bounds__(128)
void gemm_relu_wmma(const _Float16* __restrict__ A, const _Float16* __restrict__ Bt,
                    const float* __restrict__ bias, _Float16* __restrict__ C,
                    int M, int N, int K) {
    int lane = threadIdx.x & 31;
    int wave = threadIdx.x >> 5;
    int l15 = lane & 15;
    int hi = lane >> 4;
    int mt0 = blockIdx.x * 16 + wave * 4;       // first of 4 m-tiles for this wave
    int n0 = blockIdx.y * 32 + l15;             // this lane's B/C column (tile 0)
    const _Float16* arow[4];
    for (int t = 0; t < 4; ++t)
        arow[t] = A + (size_t)((mt0 + t) * 16 + l15) * K;
    const _Float16* brow0 = Bt + (size_t)n0 * K;
    const _Float16* brow1 = Bt + (size_t)(n0 + 16) * K;
    int aoff = hi ? 16 : 0;
    int boff = hi ? 16 : 0;
    v8f acc[4][2] = {};
    for (int k = 0; k < K; k += 32) {
        v16h b0 = *(const v16h*)(brow0 + k + boff);
        v16h b1 = *(const v16h*)(brow1 + k + boff);
        for (int t = 0; t < 4; ++t) {
            // prefetch next K-step of this A row (global_prefetch_b8, speculative RT)
            __builtin_prefetch(arow[t] + k + 32 + aoff, 0, 3);
            AV au;
            au.h[0] = *(const v8h*)(arow[t] + k + aoff);
            au.h[1] = *(const v8h*)(arow[t] + k + aoff + 8);
            v16h a = au.v;
            acc[t][0] = __builtin_amdgcn_wmma_f32_16x16x32_f16(false, a, false, b0, (short)0, acc[t][0], false, false);
            acc[t][1] = __builtin_amdgcn_wmma_f32_16x16x32_f16(false, a, false, b1, (short)0, acc[t][1], false, false);
        }
    }
    float bz0 = bias[n0], bz1 = bias[n0 + 16];
    for (int t = 0; t < 4; ++t) {
        for (int r = 0; r < 8; ++r) {
            int m = (mt0 + t) * 16 + r + hi * 8;
            float v0 = acc[t][0][r] + bz0; v0 = v0 > 0.f ? v0 : 0.f;
            float v1 = acc[t][1][r] + bz1; v1 = v1 > 0.f ? v1 : 0.f;
            C[(size_t)m * N + n0]      = (_Float16)v0;
            C[(size_t)m * N + n0 + 16] = (_Float16)v1;
        }
    }
}

// ---------------- detection head: 3x3x256 -> (score sigmoid, 2 reg), VALU (N=3) -----------
__global__ void det_head(const _Float16* __restrict__ FEAT,
                         const float* __restrict__ wsc, const float* __restrict__ bsc,
                         const float* __restrict__ wrg, const float* __restrict__ brg,
                         float* __restrict__ SC, float* __restrict__ REG) {
    int m = blockIdx.x * 256 + threadIdx.x;
    if (m >= BATCH * HF * WF) return;
    int x = m & 127, y = (m >> 7) & 127, b = m >> 14;
    float s = bsc[0], r0 = brg[0], r1 = brg[1];
    for (int ky = 0; ky < 3; ++ky) {
        int iy = y + ky - 1; if (iy < 0 || iy >= HF) continue;
        for (int kx = 0; kx < 3; ++kx) {
            int ix = x + kx - 1; if (ix < 0 || ix >= WF) continue;
            const _Float16* f = FEAT + (((size_t)(b * HF + iy)) * WF + ix) * 256;
            int wb = (ky * 3 + kx) * 256;
            for (int c = 0; c < 256; ++c) {
                float fv = (float)f[c];
                s  += fv * wsc[wb + c];
                r0 += fv * wrg[(wb + c) * 2];
                r1 += fv * wrg[(wb + c) * 2 + 1];
            }
        }
    }
    SC[m] = 1.f / (1.f + __expf(-s));
    REG[(size_t)m * 2]     = r0;
    REG[(size_t)m * 2 + 1] = r1;
}

// ---------------- exact top-k by rank counting (lax.top_k tie semantics) ------------------
__global__ void topk_rank(const float* __restrict__ SC, float* __restrict__ TOPS,
                          int* __restrict__ TOPI) {
    int b = blockIdx.y;
    int i = blockIdx.x * 256 + threadIdx.x;      // 0..16383
    const float* s = SC + (size_t)b * HF * WF;
    float si = s[i];
    __shared__ float sh[256];
    int rank = 0;
    for (int base = 0; base < HF * WF; base += 256) {
        sh[threadIdx.x] = s[base + threadIdx.x];
        __syncthreads();
        for (int j = 0; j < 256; ++j) {
            float sj = sh[j];
            int jj = base + j;
            rank += (sj > si) || (sj == si && jj < i);
        }
        __syncthreads();
    }
    if (rank < K_PRE) {
        TOPS[b * K_PRE + rank] = si;
        TOPI[b * K_PRE + rank] = i;
    }
}

// ---------------- decode proposal locations ----------------------------------------------
__global__ void decode_locs(const int* __restrict__ TOPI, const float* __restrict__ REG,
                            float* __restrict__ LOC) {
    int t = blockIdx.x * 256 + threadIdx.x;
    if (t >= BATCH * K_PRE) return;
    int b = t / K_PRE;
    int i = TOPI[t];
    float yy = (float)(i / WF), xx = (float)(i % WF);
    size_t ri = (size_t)b * HF * WF + i;
    LOC[(size_t)t * 2]     = yy + 0.5f + REG[ri * 2];
    LOC[(size_t)t * 2 + 1] = xx + 0.5f + REG[ri * 2 + 1];
}

// ---------------- greedy distance-NMS + select 500 + emit scores/pix + crop corners -------
__global__ void nms_select(const float* __restrict__ TOPS, const float* __restrict__ LOC,
                           float* __restrict__ out_s, float* __restrict__ out_pix,
                           int* __restrict__ TL) {
    int b = blockIdx.x;
    __shared__ float ly[K_PRE], lx[K_PRE];
    __shared__ unsigned char keep[K_PRE];
    __shared__ int conflict;
    __shared__ int sidx[K_OUT];
    __shared__ int scnt;
    for (int i = threadIdx.x; i < K_PRE; i += blockDim.x) {
        ly[i] = LOC[((size_t)b * K_PRE + i) * 2];
        lx[i] = LOC[((size_t)b * K_PRE + i) * 2 + 1];
        keep[i] = 0;
    }
    if (threadIdx.x == 0) { conflict = 0; scnt = 0; }
    __syncthreads();
    for (int i = 0; i < K_PRE; ++i) {
        float yi = ly[i], xi = lx[i];
        int c = 0;
        for (int j = threadIdx.x; j < i; j += blockDim.x) {
            if (keep[j]) {
                float dy = yi - ly[j], dx = xi - lx[j];
                if (dy * dy + dx * dx <= 1.1f * 1.1f) c = 1;
            }
        }
        if (c) atomicOr(&conflict, 1);
        __syncthreads();
        if (threadIdx.x == 0) { keep[i] = conflict ? 0 : 1; conflict = 0; }
        __syncthreads();
    }
    if (threadIdx.x == 0) {
        int cnt = 0;
        for (int i = 0; i < K_PRE; ++i)
            if (keep[i]) { if (cnt < K_OUT) sidx[cnt] = i; ++cnt; }
        scnt = cnt > K_OUT ? K_OUT : cnt;
    }
    __syncthreads();
    for (int q = threadIdx.x; q < K_OUT; q += blockDim.x) {
        float ss = -1.f, py = -1.f, px = -1.f;
        if (q < scnt) {
            int i = sidx[q];
            ss = TOPS[b * K_PRE + i];
            py = 4.f * ly[i];
            px = 4.f * lx[i];
        }
        out_s[b * K_OUT + q] = ss;
        out_pix[((size_t)b * K_OUT + q) * 2]     = py;
        out_pix[((size_t)b * K_OUT + q) * 2 + 1] = px;
        float cy = py * 0.25f, cx = px * 0.25f;   // center in feature coords
        int ty = (int)rintf(cy) - PS / 2;
        int tx = (int)rintf(cx) - PS / 2;
        ty = ty < 0 ? 0 : (ty > HF - PS ? HF - PS : ty);
        tx = tx < 0 ? 0 : (tx > WF - PS ? WF - PS : tx);
        TL[((size_t)b * K_OUT + q) * 2]     = ty;
        TL[((size_t)b * K_OUT + q) * 2 + 1] = tx;
    }
}

// ---------------- fused instance head: implicit crop + 3x3 WMMA conv + 1x1 sigmoid --------
// one block per (batch, proposal); 256 threads = 8 waves; segr stays L2-resident.
__global__ __launch_bounds__(256)
void inst_head(const _Float16* __restrict__ SEGR, const _Float16* __restrict__ BTP1,
               const float* __restrict__ bp1, const float* __restrict__ wp2,
               const float* __restrict__ bp2, const int* __restrict__ TL,
               float* __restrict__ out_inst) {
    int blk = blockIdx.x;                 // 0..1999
    int b = blk / K_OUT, q = blk % K_OUT;
    int tly = TL[(size_t)blk * 2], tlx = TL[(size_t)blk * 2 + 1];
    int lane = threadIdx.x & 31, wave = threadIdx.x >> 5;
    int l15 = lane & 15, hi = lane >> 4;
    int aoff = hi ? 16 : 0, boff = hi ? 16 : 0;
    float w2a = wp2[l15], w2b = wp2[16 + l15];
    float bz0 = bp1[l15], bz1 = bp1[16 + l15];
    float b2v = bp2[0];
    v8h z8 = {};
    for (int mg = 0; mg < 4; ++mg) {
        int mt0 = mg * 16 + wave * 2;     // two 16-pixel tiles per wave per pass
        v8f acc[2][2] = {};
        for (int tap = 0; tap < 9; ++tap) {
            int dy = tap / 3 - 1, dx = tap % 3 - 1;
            v16h b0 = *(const v16h*)(BTP1 + (size_t)l15 * 288 + tap * 32 + boff);
            v16h b1 = *(const v16h*)(BTP1 + (size_t)(16 + l15) * 288 + tap * 32 + boff);
            for (int t = 0; t < 2; ++t) {
                int P = (mt0 + t) * 16 + l15;     // pixel index within 32x32 patch
                int py = P >> 5, px = P & 31;
                int sy = py + dy, sx = px + dx;   // patch-local SAME padding
                AV au;
                if (sy >= 0 && sy < PS && sx >= 0 && sx < PS) {
                    const _Float16* src =
                        SEGR + (((size_t)(b * HF + tly + sy)) * WF + (tlx + sx)) * 32;
                    au.h[0] = *(const v8h*)(src + aoff);
                    au.h[1] = *(const v8h*)(src + aoff + 8);
                } else {
                    au.h[0] = z8; au.h[1] = z8;
                }
                v16h a = au.v;
                acc[t][0] = __builtin_amdgcn_wmma_f32_16x16x32_f16(false, a, false, b0, (short)0, acc[t][0], false, false);
                acc[t][1] = __builtin_amdgcn_wmma_f32_16x16x32_f16(false, a, false, b1, (short)0, acc[t][1], false, false);
            }
        }
        // epilogue: h = relu(acc + bp1); inst = sigmoid(h . wp2 + bp2) via 16-lane reduce
        for (int t = 0; t < 2; ++t) {
            for (int r = 0; r < 8; ++r) {
                float h0 = acc[t][0][r] + bz0; h0 = h0 > 0.f ? h0 : 0.f;
                float h1 = acc[t][1][r] + bz1; h1 = h1 > 0.f ? h1 : 0.f;
                float part = h0 * w2a + h1 * w2b;
                for (int off = 1; off < 16; off <<= 1)
                    part += __shfl_xor(part, off, 32);
                if (l15 == 0) {
                    int m = (mt0 + t) * 16 + r + hi * 8;
                    out_inst[(size_t)(b * K_OUT + q) * (PS * PS) + m] =
                        1.f / (1.f + __expf(-(part + b2v)));
                }
            }
        }
    }
}

// =========================================================================================
extern "C" void kernel_launch(void* const* d_in, const int* in_sizes, int n_in,
                              void* d_out, int out_size, void* d_ws, size_t ws_size,
                              hipStream_t stream) {
    (void)in_sizes; (void)n_in; (void)out_size; (void)ws_size;
    const float* image = (const float*)d_in[0];
    const float* w1    = (const float*)d_in[1];  const float* b1   = (const float*)d_in[2];
    const float* w2    = (const float*)d_in[3];  const float* b2   = (const float*)d_in[4];
    const float* wseg  = (const float*)d_in[5];  const float* bseg = (const float*)d_in[6];
    const float* wsc   = (const float*)d_in[7];  const float* bsc  = (const float*)d_in[8];
    const float* wrg   = (const float*)d_in[9];  const float* brg  = (const float*)d_in[10];
    const float* wr    = (const float*)d_in[11]; const float* br   = (const float*)d_in[12];
    const float* wp1   = (const float*)d_in[13]; const float* bp1  = (const float*)d_in[14];
    const float* wp2   = (const float*)d_in[15]; const float* bp2  = (const float*)d_in[16];

    char* ws = (char*)d_ws;
    size_t off = 0;
    auto carve = [&](size_t bytes) -> void* {
        void* p = ws + off;
        off = (off + bytes + 255) & ~(size_t)255;
        return p;
    };
    const size_t M1 = (size_t)BATCH * 256 * 256;   // 262144
    const size_t M2 = (size_t)BATCH * HF * WF;     // 65536

    _Float16* BT1   = (_Float16*)carve(64 * 32 * 2);
    _Float16* BT2   = (_Float16*)carve(256 * 576 * 2);
    _Float16* BTSEG = (_Float16*)carve(256 * 256 * 2);
    _Float16* BTR   = (_Float16*)carve((size_t)32 * 256 * 2);
    _Float16* BTP1  = (_Float16*)carve((size_t)32 * 288 * 2);
    _Float16* A1    = (_Float16*)carve(M1 * 32 * 2);
    _Float16* X     = (_Float16*)carve(M1 * 64 * 2);
    _Float16* A2    = (_Float16*)carve(M2 * 576 * 2);
    _Float16* FEAT  = (_Float16*)carve(M2 * 256 * 2);
    _Float16* SEG   = (_Float16*)carve(M2 * 256 * 2);
    _Float16* SEGR  = (_Float16*)carve(M2 * 32 * 2);
    float*    SC    = (float*)carve(M2 * 4);
    float*    REG   = (float*)carve(M2 * 2 * 4);
    float*    TOPS  = (float*)carve((size_t)BATCH * K_PRE * 4);
    int*      TOPI  = (int*)carve((size_t)BATCH * K_PRE * 4);
    float*    LOC   = (float*)carve((size_t)BATCH * K_PRE * 2 * 4);
    int*      TL    = (int*)carve((size_t)BATCH * K_OUT * 2 * 4);

    float* out_s    = (float*)d_out;                               // [4,500]
    float* out_pix  = out_s + BATCH * K_OUT;                       // [4,500,2]
    float* out_inst = out_pix + BATCH * K_OUT * 2;                 // [4,500,32,32,1]

    // weight repacks (N-major, K zero-padded where needed)
    repack_wT<<<(64 * 32 + 255) / 256, 256, 0, stream>>>(w1, BT1, 27, 64, 32);
    repack_wT<<<(256 * 576 + 255) / 256, 256, 0, stream>>>(w2, BT2, 576, 256, 576);
    repack_wT<<<(256 * 256 + 255) / 256, 256, 0, stream>>>(wseg, BTSEG, 256, 256, 256);
    repack_wT<<<(32 * 256 + 255) / 256, 256, 0, stream>>>(wr, BTR, 256, 32, 256);
    repack_wT<<<(32 * 288 + 255) / 256, 256, 0, stream>>>(wp1, BTP1, 288, 32, 288);

    // conv1: im2col + WMMA GEMM (M=262144, K=32, N=64)
    im2col_conv1<<<(int)((M1 + 255) / 256), 256, 0, stream>>>(image, A1);
    gemm_relu_wmma<<<dim3((unsigned)(M1 / 256), 2), 128, 0, stream>>>(A1, BT1, b1, X, (int)M1, 64, 32);

    // conv2: im2col + WMMA GEMM (M=65536, K=576, N=256)
    im2col_conv2<<<(int)((M2 * 9 + 255) / 256), 256, 0, stream>>>(X, A2);
    gemm_relu_wmma<<<dim3((unsigned)(M2 / 256), 8), 128, 0, stream>>>(A2, BT2, b2, FEAT, (int)M2, 256, 576);

    // seg (1x1) and segr (1x1) as pure WMMA GEMMs
    gemm_relu_wmma<<<dim3((unsigned)(M2 / 256), 8), 128, 0, stream>>>(FEAT, BTSEG, bseg, SEG, (int)M2, 256, 256);
    gemm_relu_wmma<<<dim3((unsigned)(M2 / 256), 1), 128, 0, stream>>>(SEG, BTR, br, SEGR, (int)M2, 32, 256);

    // detection head (N=3 -> VALU), top-k, decode, NMS + selection
    det_head<<<(int)(M2 / 256), 256, 0, stream>>>(FEAT, wsc, bsc, wrg, brg, SC, REG);
    topk_rank<<<dim3(HF * WF / 256, BATCH), 256, 0, stream>>>(SC, TOPS, TOPI);
    decode_locs<<<(BATCH * K_PRE + 255) / 256, 256, 0, stream>>>(TOPI, REG, LOC);
    nms_select<<<BATCH, 256, 0, stream>>>(TOPS, LOC, out_s, out_pix, TL);

    // fused instance head: implicit crop + 3x3 WMMA conv + 1x1 sigmoid
    inst_head<<<BATCH * K_OUT, 256, 0, stream>>>(SEGR, BTP1, bp1, wp2, bp2, TL, out_inst);
}
